// RSGCN_10505490006484
// MI455X (gfx1250) — matrix operand order repaired and checked by reading, not compiled, not run on hardware
//
#include <hip/hip_runtime.h>

// ---- problem sizes (fixed by reference) ----
#define NB   1024
#define NN   128
#define NH   64
#define NOUT 128

// LDS row strides (bf16 elements), padded to dodge bank conflicts,
// chosen so every vector access below stays 16B-aligned.
constexpr int HT_S = 136;  // hT: [64][136]  (transposed activations, bf16)
constexpr int S_S  = 72;   // S : [128][72]  (adj@h result, bf16)
constexpr int W_S  = 72;   // Wl: [128][72]  (current layer weight, bf16)

typedef __attribute__((ext_vector_type(16))) __bf16 v16bf;
typedef __attribute__((ext_vector_type(8)))  float  v8f;

union AV {
    unsigned u[8];
    uint4    q[2];
    v16bf    v;
};

// f32 -> bf16, round-to-nearest (ties away/up): +0x8000 then take high half.
// Packed pair: two integer adds + one v_perm_b32 (bytes [lo.2,lo.3,hi.2,hi.3]).
__device__ __forceinline__ unsigned pk2(float lo, float hi) {
    unsigned ulo = __builtin_bit_cast(unsigned, lo) + 0x8000u;
    unsigned uhi = __builtin_bit_cast(unsigned, hi) + 0x8000u;
    return __builtin_amdgcn_perm(uhi, ulo, 0x07060302u);
}
__device__ __forceinline__ unsigned short f2bf1(float f) {
    return (unsigned short)((__builtin_bit_cast(unsigned, f) + 0x8000u) >> 16);
}

// One GCN layer:  S = adj_b @ h   ;   X = S @ W^T  (; relu / column-sum)
// A-operand of GEMM1 (the adj strip) is pre-converted and lives in registers.
template <int OD, bool LAST>
__device__ __forceinline__ void run_layer(
    const AV (&aop)[4], const float* __restrict__ W,
    unsigned short* hT, unsigned short* S, unsigned short* Wl, float* red,
    int tid, int wave, int lrow, int lhi)
{
    // ---- stage this layer's weight into LDS as bf16 [OD][W_S] ----
    for (int e = tid; e < OD * NH; e += 256) {
        int o = e >> 6, h = e & 63;
        Wl[o * W_S + h] = f2bf1(W[e]);
    }
    __syncthreads();

    // ================= GEMM1: S = adj @ h ====================
    // wave -> n-rows [wave*16, wave*16+16); 4 h-tiles of 16 each.
    v8f acc[4];
#pragma unroll
    for (int t = 0; t < 4; ++t)
#pragma unroll
        for (int v = 0; v < 8; ++v) acc[t][v] = 0.0f;

#pragma unroll
    for (int ks = 0; ks < 4; ++ks) {
        const int kk = ks * 32;
#pragma unroll
        for (int t = 0; t < 4; ++t) {
            // B operand: hT[h][k..] two 8-bf16 chunks at k = 8*lhi and 8*lhi+16 per lane
            AV bb;
            const uint4* hp =
                (const uint4*)&hT[(t * 16 + lrow) * HT_S + kk + 8 * lhi];
            bb.q[0] = hp[0];
            bb.q[1] = hp[2];
            acc[t] = __builtin_amdgcn_wmma_f32_16x16x32_bf16(
                false, aop[ks].v, false, bb.v, (short)0, acc[t], false, false);
        }
    }

    // spill S to LDS as bf16 row-major [n][h] (transpose via b16 scatter)
#pragma unroll
    for (int t = 0; t < 4; ++t)
#pragma unroll
        for (int v = 0; v < 8; ++v)
            S[(wave * 16 + 8 * lhi + v) * S_S + t * 16 + lrow] =
                f2bf1(acc[t][v]);
    __syncthreads();

    // ================= GEMM2: X = S @ W^T ====================
    v8f acc2[OD / 16];
#pragma unroll
    for (int t = 0; t < OD / 16; ++t)
#pragma unroll
        for (int v = 0; v < 8; ++v) acc2[t][v] = 0.0f;

#pragma unroll
    for (int kk = 0; kk < NH; kk += 32) {
        AV a;
        const unsigned short* sp = &S[(wave * 16 + lrow) * S_S + kk + 8 * lhi];
        a.q[0] = *(const uint4*)sp;          // j = 0..7
        a.q[1] = *(const uint4*)(sp + 16);   // j = 8..15
#pragma unroll
        for (int t = 0; t < OD / 16; ++t) {
            AV bb;
            const uint4* wp =
                (const uint4*)&Wl[(t * 16 + lrow) * W_S + kk + 8 * lhi];
            bb.q[0] = wp[0];
            bb.q[1] = wp[2];
            acc2[t] = __builtin_amdgcn_wmma_f32_16x16x32_bf16(
                false, a.v, false, bb.v, (short)0, acc2[t], false, false);
        }
    }

    if (!LAST) {
        // relu, pack to bf16, store TRANSPOSED into hT (contiguous in n!)
#pragma unroll
        for (int t = 0; t < OD / 16; ++t) {
            AV st;
#pragma unroll
            for (int v = 0; v < 4; ++v) {
                float a0 = acc2[t][2 * v];     a0 = a0 > 0.f ? a0 : 0.f;
                float a1 = acc2[t][2 * v + 1]; a1 = a1 > 0.f ? a1 : 0.f;
                st.u[v] = pk2(a0, a1);
            }
            *(uint4*)&hT[(t * 16 + lrow) * HT_S + wave * 16 + 8 * lhi] = st.q[0];
        }
    } else {
        // readout: sum over nodes -> red[o] via LDS float atomics
#pragma unroll
        for (int t = 0; t < OD / 16; ++t) {
            float s = 0.f;
#pragma unroll
            for (int v = 0; v < 8; ++v) s += acc2[t][v];
            atomicAdd(&red[t * 16 + lrow], s);   // ds_add_f32
        }
    }
    __syncthreads();
}

__global__ void __launch_bounds__(256)
rsgcn_kernel(const int* __restrict__ graph, const float* __restrict__ adj,
             const float* __restrict__ embed, const float* __restrict__ W1,
             const float* __restrict__ W2, const float* __restrict__ W3,
             float* __restrict__ out)
{
    __shared__ __attribute__((aligned(16))) unsigned short hT[64 * HT_S];
    __shared__ __attribute__((aligned(16))) unsigned short S[128 * S_S];
    __shared__ __attribute__((aligned(16))) unsigned short Wl[128 * W_S];
    __shared__ __attribute__((aligned(16))) float red[NOUT];

    const int b    = blockIdx.x;
    const int tid  = threadIdx.x;
    const int lane = tid & 31;
    const int wave = tid >> 5;   // 0..7 (wave32)
    const int lrow = lane & 15;
    const int lhi  = lane >> 4;

    // gather node embeddings -> hT (transposed, bf16)
    const int* g = graph + b * NN;
    for (int e = tid; e < NN * NH; e += 256) {
        int n = e >> 6, h = e & 63;
        hT[h * HT_S + n] = f2bf1(embed[g[n] * NH + h]);
    }
    if (tid < NOUT) red[tid] = 0.0f;
    __syncthreads();

    // ---- convert this wave's adj strip ONCE into registers (A operands) ----
    // Per ISA 16-bit A 16x32 layout: elem j -> K = kk + 8*lhi + j (+16 for j>=8).
    // 4 k-steps x 8 VGPRs = the whole 16x128 strip; reused by all 3 layers.
    const float* arow = adj + (size_t)b * NN * NN + (size_t)(wave * 16 + lrow) * NN;
    AV aop[4];
#pragma unroll
    for (int ks = 0; ks < 4; ++ks) {
        const float4* p0 = (const float4*)(arow + ks * 32 + 8 * lhi);
        const float4* p1 = (const float4*)(arow + ks * 32 + 8 * lhi + 16);
        float4 x0 = p0[0], x1 = p0[1], y0 = p1[0], y1 = p1[1];
        aop[ks].u[0] = pk2(x0.x, x0.y); aop[ks].u[1] = pk2(x0.z, x0.w);
        aop[ks].u[2] = pk2(x1.x, x1.y); aop[ks].u[3] = pk2(x1.z, x1.w);
        aop[ks].u[4] = pk2(y0.x, y0.y); aop[ks].u[5] = pk2(y0.z, y0.w);
        aop[ks].u[6] = pk2(y1.x, y1.y); aop[ks].u[7] = pk2(y1.z, y1.w);
    }

    run_layer<64,  false>(aop, W1, hT, S, Wl, red, tid, wave, lrow, lhi);
    run_layer<64,  false>(aop, W2, hT, S, Wl, red, tid, wave, lrow, lhi);
    run_layer<128, true >(aop, W3, hT, S, Wl, red, tid, wave, lrow, lhi);

    if (tid < NOUT) out[(size_t)b * NOUT + tid] = red[tid];
}

extern "C" void kernel_launch(void* const* d_in, const int* in_sizes, int n_in,
                              void* d_out, int out_size, void* d_ws, size_t ws_size,
                              hipStream_t stream)
{
    const int*   graph = (const int*)d_in[0];
    const float* adj   = (const float*)d_in[1];
    const float* embed = (const float*)d_in[2];
    const float* W1    = (const float*)d_in[3];
    const float* W2    = (const float*)d_in[4];
    const float* W3    = (const float*)d_in[5];
    float*       out   = (float*)d_out;

    rsgcn_kernel<<<dim3(NB), dim3(256), 0, stream>>>(graph, adj, embed,
                                                     W1, W2, W3, out);
}
